// QuantumLSTMCell_65481071408893
// MI455X (gfx1250) — matrix operand order repaired and checked by reading, not compiled, not run on hardware
//
#include <hip/hip_runtime.h>

// QuantumLSTMCell for MI455X (gfx1250, wave32).
// Closed form: gate outputs are prefix products of cos(comb@W + b + th).
// GEMM (131072 x 264) @ (264 x 32) done with v_wmma_f32_16x16x32_f16,
// fp32 accumulation; ~158 MB total traffic -> bandwidth bound (~7 us @ 23.3 TB/s).

typedef __attribute__((ext_vector_type(16))) _Float16 v16h;
typedef __attribute__((ext_vector_type(8)))  _Float16 v8h;
typedef __attribute__((ext_vector_type(8)))  float    v8f;
typedef __attribute__((ext_vector_type(4)))  float    v4f;

#define KPAD  288   // 264 real K, padded to 9 * 32
#define NCOLS 32    // 4 gates x 8 qubits

// ---------------------------------------------------------------------------
// Pack kernel: Wt[n][k] (f16, n = gate*8+h, k = fan_in index, zero padded)
// and bt[n] = b[h] + theta[h] (fp32), both in workspace.
// ---------------------------------------------------------------------------
__global__ __launch_bounds__(256) void qlstm_pack(
    const float* __restrict__ Wf, const float* __restrict__ bf, const float* __restrict__ thf,
    const float* __restrict__ Wi, const float* __restrict__ bi, const float* __restrict__ thi,
    const float* __restrict__ Wu, const float* __restrict__ bu, const float* __restrict__ thu,
    const float* __restrict__ Wo, const float* __restrict__ bo, const float* __restrict__ tho,
    _Float16* __restrict__ Wt, float* __restrict__ bt)
{
    const float* Ws[4]  = {Wf, Wi, Wu, Wo};
    const float* bs[4]  = {bf, bi, bu, bo};
    const float* ths[4] = {thf, thi, thu, tho};
    for (int idx = threadIdx.x; idx < NCOLS * KPAD; idx += blockDim.x) {
        int n = idx / KPAD;
        int k = idx - n * KPAD;
        int g = n >> 3, h = n & 7;
        float v = (k < 264) ? Ws[g][k * 8 + h] : 0.0f;
        Wt[n * KPAD + k] = (_Float16)v;
    }
    if (threadIdx.x < NCOLS) {
        int g = threadIdx.x >> 3, h = threadIdx.x & 7;
        bt[threadIdx.x] = bs[g][h] + ths[g][h];
    }
}

__device__ __forceinline__ float sigmoid_f(float v) {
    return __frcp_rn(1.0f + __expf(-v));
}
__device__ __forceinline__ float tanh_f(float v) {
    // tanh(x) = 1 - 2/(exp(2x)+1)
    return 1.0f - 2.0f * __frcp_rn(1.0f + __expf(2.0f * v));
}

// ---------------------------------------------------------------------------
// Main kernel: 8 waves per WG, one 16-row tile per wave.
//   acc0: cols 0..15  (gates f, i)       acc1: cols 16..31 (gates u, o)
// ---------------------------------------------------------------------------
__global__ __launch_bounds__(256) void qlstm_main(
    const float* __restrict__ x,    // (B, 256)
    const float* __restrict__ hx,   // (B, 8)
    const float* __restrict__ cx,   // (B, 8)
    const _Float16* __restrict__ Wt, // (32, 288) f16, K-contiguous
    const float* __restrict__ bt,    // (32,)  b + theta
    float* __restrict__ out,         // [0, B*8) hx_new, [B*8, 2*B*8) cx_new
    int B)
{
    __shared__ float sAng[8][16][33];   // 33 to avoid LDS bank conflicts

    const int lane = threadIdx.x & 31;
    const int wave = threadIdx.x >> 5;
    const int m    = lane & 15;         // row-in-tile / col-in-tile
    const int hi   = lane >> 4;         // half-wave select
    const int rowBase = (blockIdx.x * 8 + wave) * 16;
    int row = rowBase + m;
    if (row >= B) row = B - 1;          // clamp loads; stores guarded below

    v8f acc0 = {};
    v8f acc1 = {};

    // A layout (16-bit 16x32): lane(m,hi) elem e (0..15) -> K = k0 + hi*16 + e
    const float* px = x + (size_t)row * 256 + hi * 16;

    #pragma unroll
    for (int ks = 0; ks < 8; ++ks) {
        const int k0 = ks * 32;
        v4f x0 = *(const v4f*)(px + k0);
        v4f x1 = *(const v4f*)(px + k0 + 4);
        v4f x2 = *(const v4f*)(px + k0 + 8);
        v4f x3 = *(const v4f*)(px + k0 + 12);
        v16h a;
        #pragma unroll
        for (int e = 0; e < 4; ++e) {
            a[e]      = (_Float16)x0[e];
            a[4 + e]  = (_Float16)x1[e];
            a[8 + e]  = (_Float16)x2[e];
            a[12 + e] = (_Float16)x3[e];
        }
        // B layout (32x16 KxN): lane = col + 16*hi, elem e -> K = k0 + hi*16 + e
        v16h b0 = *(const v16h*)(Wt + (size_t)m        * KPAD + k0 + hi * 16);
        v16h b1 = *(const v16h*)(Wt + (size_t)(16 + m) * KPAD + k0 + hi * 16);
        acc0 = __builtin_amdgcn_wmma_f32_16x16x32_f16(false, a, false, b0, (short)0, acc0, false, false);
        acc1 = __builtin_amdgcn_wmma_f32_16x16x32_f16(false, a, false, b1, (short)0, acc1, false, false);
    }

    // Tail k-step: K = 256..287 -> hx (256..263), zeros elsewhere
    {
        v16h a;
        #pragma unroll
        for (int e = 0; e < 16; ++e) a[e] = (_Float16)0.0f;
        if (hi == 0) {
            v4f h0 = *(const v4f*)(hx + (size_t)row * 8);
            v4f h1 = *(const v4f*)(hx + (size_t)row * 8 + 4);
            #pragma unroll
            for (int e = 0; e < 4; ++e) {
                a[e]     = (_Float16)h0[e];
                a[4 + e] = (_Float16)h1[e];
            }
        }
        v16h b0 = *(const v16h*)(Wt + (size_t)m        * KPAD + 256 + hi * 16);
        v16h b1 = *(const v16h*)(Wt + (size_t)(16 + m) * KPAD + 256 + hi * 16);
        acc0 = __builtin_amdgcn_wmma_f32_16x16x32_f16(false, a, false, b0, (short)0, acc0, false, false);
        acc1 = __builtin_amdgcn_wmma_f32_16x16x32_f16(false, a, false, b1, (short)0, acc1, false, false);
    }

    // D layout: lane holds col m (tile-local); VGPR r holds row r + hi*8
    #pragma unroll
    for (int r = 0; r < 8; ++r) {
        sAng[wave][hi * 8 + r][m]      = acc0[r];
        sAng[wave][hi * 8 + r][m + 16] = acc1[r];
    }
    __syncthreads();

    // Epilogue: lanes 0..15 each process one sample (row m of the tile)
    if (hi == 0) {
        const int erow = rowBase + m;
        if (erow < B) {
            float ang[32];
            #pragma unroll
            for (int j = 0; j < 32; ++j) ang[j] = sAng[wave][m][j] + bt[j];

            // Closed-form quantum gate: out_h = c0..c_h (h>=1), out_0 = c1..c7
            float go[4][8];
            #pragma unroll
            for (int g = 0; g < 4; ++g) {
                float c[8];
                #pragma unroll
                for (int h = 0; h < 8; ++h) c[h] = __cosf(ang[g * 8 + h]);
                float p = c[0];
                #pragma unroll
                for (int h = 1; h < 8; ++h) { p *= c[h]; go[g][h] = p; }
                float q = c[7];
                #pragma unroll
                for (int h = 6; h >= 1; --h) q *= c[h];
                go[g][0] = q;
            }

            #pragma unroll
            for (int h = 0; h < 8; ++h) {
                float fv = sigmoid_f(go[0][h]);
                float iv = sigmoid_f(go[1][h]);
                float gv = tanh_f(go[2][h]);
                float ov = sigmoid_f(go[3][h]);
                float cv = cx[(size_t)erow * 8 + h];
                float cnew = fv * cv + iv * gv;
                float hnew = ov * tanh_f(cnew);
                out[(size_t)erow * 8 + h] = hnew;                       // hx_new
                out[(size_t)B * 8 + (size_t)erow * 8 + h] = cnew;       // cx_new
            }
        }
    }
}

// ---------------------------------------------------------------------------
extern "C" void kernel_launch(void* const* d_in, const int* in_sizes, int n_in,
                              void* d_out, int out_size, void* d_ws, size_t ws_size,
                              hipStream_t stream)
{
    // setup_inputs() dict order:
    // 0:x 1:hx 2:cx  3:W_f 4:b_f 5:th_f  6:W_i 7:b_i 8:th_i
    // 9:W_u 10:b_u 11:th_u  12:W_o 13:b_o 14:th_o
    const float* x   = (const float*)d_in[0];
    const float* hx  = (const float*)d_in[1];
    const float* cx  = (const float*)d_in[2];
    const float* Wf  = (const float*)d_in[3];
    const float* bf  = (const float*)d_in[4];
    const float* thf = (const float*)d_in[5];
    const float* Wi  = (const float*)d_in[6];
    const float* bi  = (const float*)d_in[7];
    const float* thi = (const float*)d_in[8];
    const float* Wu  = (const float*)d_in[9];
    const float* bu  = (const float*)d_in[10];
    const float* thu = (const float*)d_in[11];
    const float* Wo  = (const float*)d_in[12];
    const float* bo  = (const float*)d_in[13];
    const float* tho = (const float*)d_in[14];

    const int B = in_sizes[0] / 256;

    _Float16* Wt = (_Float16*)d_ws;                                    // 32*288*2 = 18432 B
    float*    bt = (float*)((char*)d_ws + NCOLS * KPAD * sizeof(_Float16)); // +128 B

    qlstm_pack<<<1, 256, 0, stream>>>(Wf, bf, thf, Wi, bi, thi,
                                      Wu, bu, thu, Wo, bo, tho, Wt, bt);

    const int rowsPerWG = 8 * 16; // 8 waves * 16-row tiles
    const int nWG = (B + rowsPerWG - 1) / rowsPerWG;
    qlstm_main<<<nWG, 256, 0, stream>>>(x, hx, cx, Wt, bt, (float*)d_out, B);
}
